// Attention1_481036337164
// MI455X (gfx1250) — matrix operand order repaired and checked by reading, not compiled, not run on hardware
//
#include <hip/hip_runtime.h>
#include <hip/hip_bf16.h>
#include <stdint.h>

#define DIM    192
#define HEADS  8
#define BATCH  8
#define IMG_H  128
#define IMG_W  128
#define HW     (IMG_H * IMG_W)

typedef unsigned short u16;
typedef __bf16 v16bf __attribute__((ext_vector_type(16)));
typedef float  v8f   __attribute__((ext_vector_type(8)));
typedef unsigned int u32x4 __attribute__((ext_vector_type(4)));
typedef int          i32x8 __attribute__((ext_vector_type(8)));
typedef int          i32x4 __attribute__((ext_vector_type(4)));

#if defined(__has_builtin)
#if __has_builtin(__builtin_amdgcn_tensor_load_to_lds) && \
    __has_builtin(__builtin_amdgcn_s_wait_tensorcnt)
#define USE_TDM 1
#endif
#endif

static __device__ __forceinline__ u16 f32_to_bf16_rne(float f) {
    unsigned int u = __float_as_uint(f);
    unsigned int r = u + 0x7FFFu + ((u >> 16) & 1u);
    return (u16)(r >> 16);
}

union BFrag {
    u16   u[16];
    uint4 q[2];
    v16bf v;
};

#ifdef USE_TDM
// --------------------------------------------------------------------------
// TDM: async 2D tile load global -> LDS (f32 elements).
// D# built per CDNA5 ISA (group0) / §8.4 (group1).
// rows x cols tile, row stride in elements; LDS destination contiguous.
// This toolchain uses the 6-arg builtin: (g0, g1, g2, g3, g4, cpol).
// --------------------------------------------------------------------------
static __device__ __forceinline__ void tdm_load_2d_f32(
    const float* gsrc, unsigned lds_off_bytes,
    unsigned rows, unsigned cols, unsigned row_stride_elems)
{
    const unsigned long long ga = (unsigned long long)(uintptr_t)gsrc;
    u32x4 g0;
    g0[0] = 1u;                                   // count=1, user descriptor
    g0[1] = lds_off_bytes;                        // lds_addr (bytes)
    g0[2] = (unsigned)(ga & 0xFFFFFFFFull);       // global_addr[31:0]
    g0[3] = (unsigned)((ga >> 32) & 0x01FFFFFFull) | (2u << 30); // [56:32]|type=2
    i32x8 g1;
    g1[0] = (int)(2u << 16);                      // data_size = 4 bytes
    g1[1] = (int)((row_stride_elems & 0xFFFFu) << 16);           // tensor_dim0[15:0]
    g1[2] = (int)((row_stride_elems >> 16) | (rows << 16));      // dim0[31:16]|dim1[15:0]
    g1[3] = (int)(cols << 16);                    // dim1[31:16]=0 | tile_dim0
    g1[4] = (int)rows;                            // tile_dim1 | tile_dim2=0
    g1[5] = (int)row_stride_elems;                // tensor_dim0_stride[31:0]
    g1[6] = 0;                                    // stride[47:32] | dim1_stride[15:0]
    g1[7] = 0;                                    // dim1_stride[47:16]
    const i32x4 z4 = {0, 0, 0, 0};                // groups 2/3 unused (2D tile)
    const i32x8 z8 = {0, 0, 0, 0, 0, 0, 0, 0};    // extra group (unused)
    __builtin_amdgcn_tensor_load_to_lds(g0, g1, z4, z4, z8, 0);
}
#endif

// --------------------------------------------------------------------------
// Elementwise f32 -> bf16 (vectorized 8/thread). n must be a multiple of 8.
// --------------------------------------------------------------------------
__global__ __launch_bounds__(256)
void cvt_f32_bf16(const float* __restrict__ in, u16* __restrict__ out, size_t n)
{
    const size_t i = ((size_t)blockIdx.x * blockDim.x + threadIdx.x) * 8;
    if (i >= n) return;
    const float4 f0 = *(const float4*)(in + i);
    const float4 f1 = *(const float4*)(in + i + 4);
    u16 o[8];
    o[0] = f32_to_bf16_rne(f0.x); o[1] = f32_to_bf16_rne(f0.y);
    o[2] = f32_to_bf16_rne(f0.z); o[3] = f32_to_bf16_rne(f0.w);
    o[4] = f32_to_bf16_rne(f1.x); o[5] = f32_to_bf16_rne(f1.y);
    o[6] = f32_to_bf16_rne(f1.z); o[7] = f32_to_bf16_rne(f1.w);
    *(uint4*)(out + i) = *(const uint4*)o;
}

// --------------------------------------------------------------------------
// Weight repack: src f32 [M][K][taps] -> dst bf16 [M][taps][K]
// --------------------------------------------------------------------------
__global__ __launch_bounds__(256)
void repack_w(const float* __restrict__ src, u16* __restrict__ dst,
              int M, int K, int taps)
{
    const int i = blockIdx.x * 256 + threadIdx.x;
    if (i >= M * K * taps) return;
    const int t = i % taps;
    const int k = (i / taps) % K;
    const int m = i / (taps * K);
    dst[((size_t)m * taps + t) * K + k] = f32_to_bf16_rne(src[i]);
}

// --------------------------------------------------------------------------
// Implicit-GEMM conv (taps==1: 1x1; taps==9: full 3x3 SAME).
// in  : bf16 [B][K][HW]      wgt : bf16 [M][taps][K]
// out : f32 or bf16 [B][M][HW]
// WG tile 64(M) x 128(N) (one image row). 8 waves, each 16(M) x 64(N).
// --------------------------------------------------------------------------
__global__ __launch_bounds__(256)
void gemm_conv_wmma(const u16* __restrict__ in, const u16* __restrict__ wgt,
                    void* __restrict__ out, int M, int K, int taps, int out_bf16)
{
    __shared__ __align__(16) u16 ldsb[8 * 32 * 16];   // [jt][lane][e]

    const int tid  = threadIdx.x;
    const int lane = tid & 31;
    const int wid  = tid >> 5;
    const int wm   = wid & 3;              // 4 waves along M
    const int wn   = wid >> 2;             // 2 waves along N
    const int half = lane >> 4;
    const int l15  = lane & 15;

    const int b     = blockIdx.z;
    const int mbase = blockIdx.y * 64 + wm * 16;
    const int pbase = blockIdx.x * 128;
    const int ybase = pbase >> 7;          // tile == one image row (IMG_W==128)

    v8f acc[4];
#pragma unroll
    for (int j = 0; j < 4; ++j)
#pragma unroll
        for (int r = 0; r < 8; ++r) acc[j][r] = 0.f;

    const u16* inB = in + (size_t)b * K * HW;

    // B-tile writer mapping: thread = (k-pair a, 8-col group oct); 16 values each
    const int a   = tid >> 4;              // 0..15 -> k = 2a, 2a+1
    const int oct = tid & 15;              // 0..15 -> n = oct*8 .. +7
    const int n0l = oct * 8;
    const int wjt = n0l >> 4;              // fragment-tile index
    const int wl  = (n0l & 15) + ((a >> 3) << 4);  // lane base (+16 when k>=16)
    const int we  = (2 * a) & 15;          // even element index

    for (int t = 0; t < taps; ++t) {
        int ty = 0, tx = 0;
        if (taps > 1) { ty = t / 3 - 1; tx = t % 3 - 1; }

        for (int k0 = 0; k0 < K; k0 += 32) {
            // ---- stage B tile (32k x 128n bf16) into fragment-layout LDS ----
            {
                const int  cin = k0 + 2 * a;
                const u16* s0  = inB + (size_t)cin * HW;
                const u16* s1  = s0 + HW;
                if (cin + 32 < K)
                    __builtin_prefetch(s0 + 32 * HW + ybase * IMG_W, 0, 1);

                u16 lo[8], hi[8];
                const int yy = ybase + ty;
                if (yy >= 0 && yy < IMG_H) {
                    const u16* r0 = s0 + yy * IMG_W;
                    const u16* r1 = s1 + yy * IMG_W;
                    const int  xs = n0l + tx;
                    if (xs >= 0 && xs + 7 < IMG_W) {
                        if ((xs & 7) == 0) {             // aligned vector path (tx==0)
                            *(uint4*)lo = *(const uint4*)(r0 + xs);
                            *(uint4*)hi = *(const uint4*)(r1 + xs);
                        } else {
#pragma unroll
                            for (int j = 0; j < 8; ++j) { lo[j] = r0[xs + j]; hi[j] = r1[xs + j]; }
                        }
                    } else {                              // row-edge path
#pragma unroll
                        for (int j = 0; j < 8; ++j) {
                            const int xx = xs + j;
                            const bool ok = (xx >= 0) && (xx < IMG_W);
                            lo[j] = ok ? r0[xx] : (u16)0;
                            hi[j] = ok ? r1[xx] : (u16)0;
                        }
                    }
                } else {
#pragma unroll
                    for (int j = 0; j < 8; ++j) { lo[j] = 0; hi[j] = 0; }
                }
#pragma unroll
                for (int j = 0; j < 8; ++j) {
                    const unsigned pk = (unsigned)lo[j] | ((unsigned)hi[j] << 16);
                    *(unsigned*)&ldsb[((wjt * 32) + wl + j) * 16 + we] = pk;
                }
            }
            __syncthreads();

            // ---- A fragment: two aligned 16B bf16 loads per lane ----
            BFrag afr;
            {
                const u16* wrow = wgt + ((size_t)(mbase + l15) * taps + t) * K
                                      + k0 + 16 * half;
                afr.q[0] = *(const uint4*)(wrow);        // e0..7  -> k0 + 16h + 0..7
                afr.q[1] = *(const uint4*)(wrow + 8);    // e8..15 -> k0 + 16h + 8..15
            }

            // ---- 4 B fragments (contiguous 32B ds reads) + 4 WMMAs ----
#pragma unroll
            for (int jj = 0; jj < 4; ++jj) {
                const int jt = wn * 4 + jj;
                BFrag bfr;
                const uint4* p = (const uint4*)&ldsb[(jt * 32 + lane) * 16];
                bfr.q[0] = p[0];
                bfr.q[1] = p[1];
                acc[jj] = __builtin_amdgcn_wmma_f32_16x16x32_bf16(
                    false, afr.v, false, bfr.v, (short)0, acc[jj], false, false);
            }
            __syncthreads();
        }
    }

    // ---- writeback: D layout VGPR r -> M = r + 8*half, N = lane&15 ----
    if (!out_bf16) {
        float* ob = (float*)out;
#pragma unroll
        for (int jj = 0; jj < 4; ++jj) {
            const int n = pbase + (wn * 4 + jj) * 16 + l15;
#pragma unroll
            for (int r = 0; r < 8; ++r) {
                const int m = mbase + r + 8 * half;
                ob[((size_t)b * M + m) * HW + n] = acc[jj][r];
            }
        }
    } else {
        u16* ob = (u16*)out;
#pragma unroll
        for (int jj = 0; jj < 4; ++jj) {
            const int n = pbase + (wn * 4 + jj) * 16 + l15;
#pragma unroll
            for (int r = 0; r < 8; ++r) {
                const int m = mbase + r + 8 * half;
                ob[((size_t)b * M + m) * HW + n] = f32_to_bf16_rne(acc[jj][r]);
            }
        }
    }
}

// --------------------------------------------------------------------------
// Depthwise 3x3 SAME conv (groups == channels), f32, memory bound.
// --------------------------------------------------------------------------
__global__ __launch_bounds__(256)
void dwconv3x3(const float* __restrict__ in, const float* __restrict__ w,
               float* __restrict__ out, int C)
{
    const size_t idx   = (size_t)blockIdx.x * blockDim.x + threadIdx.x;
    const size_t total = (size_t)BATCH * C * HW;
    if (idx >= total) return;
    const int p  = (int)(idx % HW);
    const int ch = (int)((idx / HW) % C);
    const int y  = p / IMG_W, x = p % IMG_W;
    const float* src = in + (idx - (size_t)p);
    const float* wt  = w + ch * 9;
    float s = 0.f;
#pragma unroll
    for (int t = 0; t < 9; ++t) {
        const int yy = y + t / 3 - 1;
        const int xx = x + t % 3 - 1;
        if (yy >= 0 && yy < IMG_H && xx >= 0 && xx < IMG_W)
            s += wt[t] * src[yy * IMG_W + xx];
    }
    out[idx] = s;
}

// --------------------------------------------------------------------------
// Fused: spatial L2 norms + q.k^T (24x24) + temperature + softmax.
// One block per (b, head); 576 threads: thread (c,d) owns one dot product.
// q/k chunk slabs staged via the Tensor Data Mover (wave 0 issues two D#s,
// waits TENSORcnt, then the workgroup barrier publishes the LDS tiles).
// --------------------------------------------------------------------------
#define CHUNK 128
__global__ __launch_bounds__(576)
void attn_softmax(const float* __restrict__ q, const float* __restrict__ kv,
                  const float* __restrict__ temp, float* __restrict__ attn)
{
    __shared__ __align__(16) float qs[24][CHUNK];
    __shared__ __align__(16) float ks[24][CHUNK];
    __shared__ float sm[24 * 24];
    __shared__ float qq[24], kk_[24];

    const int tid = threadIdx.x;
    const int bh  = blockIdx.x;
    const int b   = bh / HEADS, h = bh % HEADS;
    const int c   = tid / 24, d = tid % 24;

    const float* qb = q  + ((size_t)b * DIM + h * 24) * HW;
    const float* kb = kv + ((size_t)b * 2 * DIM + h * 24) * HW;   // k = first C chans

    float dot = 0.f, sq = 0.f, sk = 0.f;
    for (int n0 = 0; n0 < HW; n0 += CHUNK) {
#ifdef USE_TDM
        if (tid < 32) {                      // wave 0 drives the TDM
            tdm_load_2d_f32(qb + n0, (unsigned)(uintptr_t)&qs[0][0], 24, CHUNK, HW);
            tdm_load_2d_f32(kb + n0, (unsigned)(uintptr_t)&ks[0][0], 24, CHUNK, HW);
            __builtin_amdgcn_s_wait_tensorcnt(0);
        }
        __syncthreads();
#else
        for (int i = tid; i < 24 * CHUNK; i += 576) {
            const int cc = i / CHUNK, j = i % CHUNK;
            qs[cc][j] = qb[(size_t)cc * HW + n0 + j];
            ks[cc][j] = kb[(size_t)cc * HW + n0 + j];
        }
        __syncthreads();
#endif
#pragma unroll 4
        for (int j = 0; j < CHUNK; ++j) {
            const float av = qs[c][j], bv = ks[d][j];
            dot += av * bv;
            if (c == d) { sq += av * av; sk += bv * bv; }
        }
        __syncthreads();
    }
    if (c == d) { qq[c] = sq; kk_[c] = sk; }
    __syncthreads();

    const float qn = fmaxf(sqrtf(qq[c]),  1e-12f);
    const float kn = fmaxf(sqrtf(kk_[d]), 1e-12f);
    sm[c * 24 + d] = dot / (qn * kn) * temp[h];
    __syncthreads();

    if (tid < 24) {
        float mx = -1e30f;
        for (int j = 0; j < 24; ++j) mx = fmaxf(mx, sm[tid * 24 + j]);
        float e[24], s = 0.f;
        for (int j = 0; j < 24; ++j) { e[j] = __expf(sm[tid * 24 + j] - mx); s += e[j]; }
        const float inv = 1.f / s;
        float* dst = attn + ((size_t)bh * 24 + tid) * 24;
        for (int j = 0; j < 24; ++j) dst[j] = e[j] * inv;
    }
}

// --------------------------------------------------------------------------
// outa_bf16[b, h*24+c, p] = bf16( sum_d attn[b,h,c,d] * v[b, h*24+d, p] )
// --------------------------------------------------------------------------
__global__ __launch_bounds__(256)
void apply_attn_v(const float* __restrict__ attn, const float* __restrict__ kv,
                  u16* __restrict__ outa)
{
    __shared__ float at[24 * 24];
    const int tid = threadIdx.x;
    const int h = blockIdx.y, b = blockIdx.z;
    const int p = blockIdx.x * 256 + tid;

    const float* asrc = attn + (size_t)(b * HEADS + h) * 24 * 24;
    for (int i = tid; i < 576; i += 256) at[i] = asrc[i];
    __syncthreads();

    const float* vb = kv + ((size_t)b * 2 * DIM + DIM + h * 24) * HW + p;
    float vv[24];
#pragma unroll
    for (int d = 0; d < 24; ++d) vv[d] = vb[(size_t)d * HW];

    u16* ob = outa + ((size_t)b * DIM + h * 24) * HW + p;
#pragma unroll
    for (int c = 0; c < 24; ++c) {
        float s = 0.f;
#pragma unroll
        for (int d = 0; d < 24; ++d) s += at[c * 24 + d] * vv[d];
        ob[(size_t)c * HW] = f32_to_bf16_rne(s);
    }
}

// --------------------------------------------------------------------------
extern "C" void kernel_launch(void* const* d_in, const int* in_sizes, int n_in,
                              void* d_out, int out_size, void* d_ws, size_t ws_size,
                              hipStream_t stream)
{
    const float* x      = (const float*)d_in[0];
    const float* y      = (const float*)d_in[1];
    const float* kv_w   = (const float*)d_in[2];
    const float* kv_dw  = (const float*)d_in[3];
    const float* q_w    = (const float*)d_in[4];
    const float* q_dw   = (const float*)d_in[5];
    const float* proj_w = (const float*)d_in[6];
    const float* temp   = (const float*)d_in[7];
    float* out = (float*)d_out;

    char* ws = (char*)d_ws;
    const size_t S2 = (size_t)BATCH * 2 * DIM * HW * sizeof(float);  // 201 MB
    const size_t S1 = (size_t)BATCH * DIM * HW * sizeof(float);      // 101 MB
    // region 0 [0, S2):        kv_raw f32; later aliased by qmid_bf / outa_bf
    // region 1 [S2, 2*S2):     kv_buf f32 (k|v), live to the end
    // region 2 [2*S2, +S1):    xbf + ybf (prep), later aliased by qfull f32
    float* kv_raw  = (float*)(ws);
    u16*   qmid_bf = (u16*)(ws);                    // dead region reuse
    u16*   outa_bf = (u16*)(ws);                    // dead region reuse
    float* kv_buf  = (float*)(ws + S2);
    u16*   xbf     = (u16*)(ws + 2 * S2);
    u16*   ybf     = (u16*)(ws + 2 * S2 + S1 / 2);
    float* qfull   = (float*)(ws + 2 * S2);         // overwrites xbf/ybf (dead)
    size_t off     = 2 * S2 + S1;
    float* attn    = (float*)(ws + off);  off += ((size_t)BATCH * HEADS * 24 * 24 * 4 + 255) & ~(size_t)255;
    u16*   wkv_bf  = (u16*)(ws + off);    off += ((size_t)2 * DIM * DIM * 2 + 255) & ~(size_t)255;
    u16*   wq_bf   = (u16*)(ws + off);    off += ((size_t)DIM * DIM * 2 + 255) & ~(size_t)255;
    u16*   wqdw_bf = (u16*)(ws + off);    off += ((size_t)DIM * DIM * 9 * 2 + 255) & ~(size_t)255;
    u16*   wproj_bf= (u16*)(ws + off);

    const dim3 blk(256);
    const size_t nElem = (size_t)BATCH * DIM * HW;

    // 0) prep: inputs -> bf16; weights -> bf16 [M][taps][K]
    cvt_f32_bf16<<<dim3((unsigned)(nElem / 8 / 256)), blk, 0, stream>>>(x, xbf, nElem);
    cvt_f32_bf16<<<dim3((unsigned)(nElem / 8 / 256)), blk, 0, stream>>>(y, ybf, nElem);
    repack_w<<<dim3((2 * DIM * DIM + 255) / 256), blk, 0, stream>>>(kv_w,  wkv_bf,  2 * DIM, DIM, 1);
    repack_w<<<dim3((DIM * DIM + 255) / 256),     blk, 0, stream>>>(q_w,   wq_bf,   DIM, DIM, 1);
    repack_w<<<dim3((DIM * DIM * 9 + 255) / 256), blk, 0, stream>>>(q_dw,  wqdw_bf, DIM, DIM, 9);
    repack_w<<<dim3((DIM * DIM + 255) / 256),     blk, 0, stream>>>(proj_w,wproj_bf,DIM, DIM, 1);

    // 1) kv_raw = kv_w(384x192) @ x           (bf16 WMMA GEMM -> f32)
    gemm_conv_wmma<<<dim3(HW / 128, (2 * DIM) / 64, BATCH), blk, 0, stream>>>(
        xbf, wkv_bf, kv_raw, 2 * DIM, DIM, 1, 0);

    // 2) kv_buf = depthwise3x3(kv_raw)
    {
        const size_t total = (size_t)BATCH * 2 * DIM * HW;
        dwconv3x3<<<dim3((unsigned)((total + 255) / 256)), blk, 0, stream>>>(
            kv_raw, kv_dw, kv_buf, 2 * DIM);
    }

    // 3) qmid_bf = q_w @ y                    (WMMA GEMM -> bf16, reuses region 0)
    gemm_conv_wmma<<<dim3(HW / 128, DIM / 64, BATCH), blk, 0, stream>>>(
        ybf, wq_bf, qmid_bf, DIM, DIM, 1, 1);

    // 4) qfull = conv3x3(qmid_bf, q_dw)       (implicit GEMM, 9 taps, dominant 87 GF)
    gemm_conv_wmma<<<dim3(HW / 128, DIM / 64, BATCH), blk, 0, stream>>>(
        qmid_bf, wqdw_bf, qfull, DIM, DIM, 9, 0);

    // 5) attn = softmax(norm(q) . norm(k)^T * temperature)   [TDM-staged]
    attn_softmax<<<dim3(BATCH * HEADS), dim3(576), 0, stream>>>(
        qfull, kv_buf, temp, attn);

    // 6) outa_bf = attn @ v                   (bf16 out, reuses region 0)
    apply_attn_v<<<dim3(HW / 256, HEADS, BATCH), blk, 0, stream>>>(
        attn, kv_buf, outa_bf);

    // 7) out = proj_w @ outa                  (bf16 WMMA GEMM -> f32)
    gemm_conv_wmma<<<dim3(HW / 128, DIM / 64, BATCH), blk, 0, stream>>>(
        outa_bf, wproj_bf, out, DIM, DIM, 1, 0);
}
